// CrossAttention_43267500540561
// MI455X (gfx1250) — matrix operand order repaired and checked by reading, not compiled, not run on hardware
//
#include <hip/hip_runtime.h>
#include <hip/hip_bf16.h>
#include <stdint.h>

// ---------------------------------------------------------------------------
// Problem constants (match reference)
// ---------------------------------------------------------------------------
#define BB      2
#define SEQ     2048
#define DMODEL  1024
#define H_HEADS 16
#define DHD     64

typedef __attribute__((ext_vector_type(16))) _Float16 v16h;
typedef __attribute__((ext_vector_type(8)))  float    v8f;

union Frag16 { v16h v; unsigned u[8]; };
union H2     { _Float16 h[2]; unsigned u; };
union H1     { _Float16 h; unsigned short s; };

static __device__ inline v8f zero8() {
    v8f z = {0.f,0.f,0.f,0.f,0.f,0.f,0.f,0.f};
    return z;
}

// ---------------------------------------------------------------------------
// gfx1250 async global->LDS copy (ASYNCcnt-tracked), with safe fallback
// ---------------------------------------------------------------------------
#define AS1 __attribute__((address_space(1)))
#define AS3 __attribute__((address_space(3)))

typedef int v4i_vs __attribute__((vector_size(16)));   // matches builtin param type

#if defined(__gfx1250__) && __has_builtin(__builtin_amdgcn_global_load_async_to_lds_b128)
#define HAVE_ASYNC_LDS 1
#else
#define HAVE_ASYNC_LDS 0
#endif

// Copy 16 contiguous bytes global -> LDS (per lane).
static __device__ inline void copy16_g2l(unsigned short* lds_dst, const _Float16* gsrc) {
#if HAVE_ASYNC_LDS
    __builtin_amdgcn_global_load_async_to_lds_b128(
        (AS1 v4i_vs*)gsrc, (AS3 v4i_vs*)lds_dst, 0, 0);
#else
    *(uint4*)lds_dst = *(const uint4*)gsrc;
#endif
}

static __device__ inline void wait_g2l() {
#if HAVE_ASYNC_LDS
#if __has_builtin(__builtin_amdgcn_s_wait_asynccnt)
    __builtin_amdgcn_s_wait_asynccnt(0);
#else
    asm volatile("s_wait_asynccnt 0x0" ::: "memory");
#endif
#endif
}

// ---------------------------------------------------------------------------
// f32 -> f16 bulk convert (vectorized, one-time cost per tensor)
// ---------------------------------------------------------------------------
__global__ __launch_bounds__(256)
void cvt_f32_to_f16(const float* __restrict__ src, _Float16* __restrict__ dst, int n4)
{
    int i = blockIdx.x * blockDim.x + threadIdx.x;
    if (i >= n4) return;
    float4 v = ((const float4*)src)[i];
    H2 p0, p1;
    p0.h[0] = (_Float16)v.x; p0.h[1] = (_Float16)v.y;
    p1.h[0] = (_Float16)v.z; p1.h[1] = (_Float16)v.w;
    ((uint2*)dst)[i] = make_uint2(p0.u, p1.u);
}

// ---------------------------------------------------------------------------
// WMMA GEMM:  C[M][N] = Xh[M][K] * Wh[N][K]^T + bias[N]   (f16 in, f32 out)
// Block tile 128x64, BK=32, 128 threads (4 waves); each wave: 32 rows x 64
// cols = 2x4 WMMA subtiles per k-step. Double-buffered async global->LDS
// staging: tile k+1 streams in (ASYNCcnt) while tile k is consumed by WMMA.
// ---------------------------------------------------------------------------
#define GBM 128
#define GBN 64
#define GBK 32
#define LDT 40   // halves per LDS tile row (32 data + 8 pad); row = 80B, 16B-aligned

__global__ __launch_bounds__(128)
void gemm_h_xwT_bias(const _Float16* __restrict__ Xh, const _Float16* __restrict__ Wh,
                     const float* __restrict__ bias, float* __restrict__ C,
                     int M, int N, int K)
{
    __shared__ __align__(16) unsigned short sA[2][GBM * LDT];
    __shared__ __align__(16) unsigned short sB[2][GBN * LDT];

    const int tid  = threadIdx.x;
    const int wave = tid >> 5;
    const int lane = tid & 31;
    const int lm   = lane & 15;
    const int lh   = lane >> 4;
    const int tileM = blockIdx.y * GBM;
    const int tileN = blockIdx.x * GBN;

    auto stage = [&](int buf, int kk) {
#if HAVE_ASYNC_LDS
        #pragma unroll
        for (int i = 0; i < 4; ++i) {            // A: 512 x 16B chunks
            int idx = tid + i * 128;
            int row = idx >> 2, ch = (idx & 3) << 3;
            copy16_g2l(&sA[buf][row * LDT + ch], Xh + (size_t)(tileM + row) * K + kk + ch);
        }
        #pragma unroll
        for (int i = 0; i < 2; ++i) {            // B: 256 x 16B chunks
            int idx = tid + i * 128;
            int row = idx >> 2, ch = (idx & 3) << 3;
            copy16_g2l(&sB[buf][row * LDT + ch], Wh + (size_t)(tileN + row) * K + kk + ch);
        }
#else
        uint4 ra[4], rb[2];
        #pragma unroll
        for (int i = 0; i < 4; ++i) {
            int idx = tid + i * 128;
            int row = idx >> 2, ch = (idx & 3) << 3;
            ra[i] = *(const uint4*)(Xh + (size_t)(tileM + row) * K + kk + ch);
        }
        #pragma unroll
        for (int i = 0; i < 2; ++i) {
            int idx = tid + i * 128;
            int row = idx >> 2, ch = (idx & 3) << 3;
            rb[i] = *(const uint4*)(Wh + (size_t)(tileN + row) * K + kk + ch);
        }
        #pragma unroll
        for (int i = 0; i < 4; ++i) {
            int idx = tid + i * 128;
            int row = idx >> 2, ch = (idx & 3) << 3;
            *(uint4*)&sA[buf][row * LDT + ch] = ra[i];
        }
        #pragma unroll
        for (int i = 0; i < 2; ++i) {
            int idx = tid + i * 128;
            int row = idx >> 2, ch = (idx & 3) << 3;
            *(uint4*)&sB[buf][row * LDT + ch] = rb[i];
        }
#endif
    };

    v8f acc[2][4];
    for (int mi = 0; mi < 2; ++mi)
        for (int nt = 0; nt < 4; ++nt) acc[mi][nt] = zero8();

    // prologue: stage first tile
    stage(0, 0);
    wait_g2l();
    __syncthreads();

    int buf = 0;
    for (int kk = 0; kk < K; kk += GBK) {
        // kick off next tile into the other buffer (overlaps with WMMA below)
        if (kk + GBK < K) {
            stage(buf ^ 1, kk + GBK);
            if (kk + 2 * GBK < K) {
                __builtin_prefetch(Xh + (size_t)(tileM + tid) * K + kk + 2 * GBK, 0, 3);
                __builtin_prefetch(Wh + (size_t)(tileN + (tid >> 1)) * K + kk + 2 * GBK, 0, 3);
            }
        }

        // ---- A fragments (2 m-subtiles), ISA 16-bit A layout ----
        Frag16 a[2];
        #pragma unroll
        for (int mi = 0; mi < 2; ++mi)
            #pragma unroll
            for (int v = 0; v < 8; ++v) {
                int k = ((v >> 2) << 4) + (lh << 3) + ((v & 3) << 1);
                a[mi].u[v] = *(const unsigned*)&sA[buf][(wave * 32 + mi * 16 + lm) * LDT + k];
            }
        // ---- B fragments + 8 WMMAs ----
        #pragma unroll
        for (int nt = 0; nt < 4; ++nt) {
            Frag16 b;
            #pragma unroll
            for (int v = 0; v < 8; ++v) {
                int k = ((v >> 2) << 4) + (lh << 3) + ((v & 3) << 1);
                b.u[v] = *(const unsigned*)&sB[buf][(nt * 16 + lm) * LDT + k];
            }
            #pragma unroll
            for (int mi = 0; mi < 2; ++mi)
                acc[mi][nt] = __builtin_amdgcn_wmma_f32_16x16x32_f16(
                    false, a[mi].v, false, b.v, (short)0, acc[mi][nt], false, false);
        }

        wait_g2l();          // next tile landed in LDS
        __syncthreads();     // all waves done reading buf / writes visible
        buf ^= 1;
    }

    // ---- epilogue ----
    for (int nt = 0; nt < 4; ++nt) {
        int n = tileN + nt * 16 + lm;
        float bv = bias[n];
        #pragma unroll
        for (int mi = 0; mi < 2; ++mi)
            #pragma unroll
            for (int v = 0; v < 8; ++v) {
                int m = tileM + wave * 32 + mi * 16 + v + (lh << 3);
                C[(size_t)m * N + n] = acc[mi][nt][v] + bv;
            }
    }
}

// ---------------------------------------------------------------------------
// RMSNorm (full D) + RoPE, output f16 per head: Oh[B][H][S][DH]
// ---------------------------------------------------------------------------
__global__ __launch_bounds__(256)
void rmsnorm_rope_pack(const float* __restrict__ T, const float* __restrict__ Wn,
                       const float* __restrict__ Pe, _Float16* __restrict__ Oh)
{
    __shared__ float red[8];
    const int row = blockIdx.x;           // b*SEQ + s
    const int b = row / SEQ;
    const int s = row % SEQ;
    const int tid = threadIdx.x;

    const float4 xv = ((const float4*)(T + (size_t)row * DMODEL))[tid];
    float ss = xv.x*xv.x + xv.y*xv.y + xv.z*xv.z + xv.w*xv.w;
    #pragma unroll
    for (int off = 16; off > 0; off >>= 1) ss += __shfl_xor(ss, off);
    if ((tid & 31) == 0) red[tid >> 5] = ss;
    __syncthreads();
    float tot = 0.f;
    #pragma unroll
    for (int i = 0; i < 8; ++i) tot += red[i];
    const float rinv = rsqrtf(tot / (float)DMODEL + 1e-5f);

    #pragma unroll
    for (int jj = 0; jj < 2; ++jj) {
        int j  = tid + jj * 256;          // pair index 0..511
        int d0 = 2 * j;
        float t0 = T[(size_t)row * DMODEL + d0]     * rinv * Wn[d0];
        float t1 = T[(size_t)row * DMODEL + d0 + 1] * rinv * Wn[d0 + 1];
        float4 pv = *(const float4*)(Pe + ((size_t)s * (DMODEL / 2) + j) * 4);
        float o0 = pv.x * t0 + pv.y * t1;
        float o1 = pv.z * t0 + pv.w * t1;
        int h  = d0 >> 6;
        int dh = d0 & 63;
        size_t base = ((size_t)(b * H_HEADS + h) * SEQ + s) * DHD + dh;
        Oh[base]     = (_Float16)o0;
        Oh[base + 1] = (_Float16)o1;
    }
}

// ---------------------------------------------------------------------------
// Pack V (f32, [B][S][D]) -> f16 transposed per head: Vt[B][H][DH][S]
// ---------------------------------------------------------------------------
__global__ __launch_bounds__(256)
void pack_v_t(const float* __restrict__ V, _Float16* __restrict__ Vt, int total)
{
    int idx = blockIdx.x * blockDim.x + threadIdx.x;
    if (idx >= total) return;
    int e  = idx % DMODEL;
    int rs = idx / DMODEL;
    int s  = rs % SEQ;
    int b  = rs / SEQ;
    int h  = e >> 6;
    int dh = e & 63;
    Vt[((size_t)(b * H_HEADS + h) * DHD + dh) * SEQ + s] = (_Float16)V[idx];
}

// ---------------------------------------------------------------------------
// Flash-style attention. Block = 4 waves sharing one (b,h); each wave owns a
// 16-query tile. K/V tiles for each 32-kv step are staged ONCE per block via
// double-buffered async global->LDS (next tile streams in while this one is
// consumed). Output written as f16 [B][S][H*DH] (A-operand of final GEMM).
// ---------------------------------------------------------------------------
#define LDKK 72   // halves per K-tile row (64 data + 8 pad); 144B, 16B-aligned
#define LDVK 40   // halves per V-tile row (32 data + 8 pad); 80B, 16B-aligned
#define LPP  36   // halves per P-row in per-wave transpose slab

__global__ __launch_bounds__(128)
void attn_fwd(const _Float16* __restrict__ Qh, const _Float16* __restrict__ Kh,
              const _Float16* __restrict__ Vt, _Float16* __restrict__ Out)
{
    __shared__ __align__(16) unsigned short sK[2][32 * LDKK];  // [kv][dh]
    __shared__ __align__(16) unsigned short sV[2][64 * LDVK];  // [dh][kv]
    __shared__ __align__(8)  unsigned short sP[4][16 * LPP];

    const int tid   = threadIdx.x;
    const int lane  = tid & 31;
    const int wave  = tid >> 5;
    const int lm    = lane & 15;
    const int lh    = lane >> 4;
    const int qbase = blockIdx.x * 64 + wave * 16;
    const int hh    = blockIdx.y;
    const int b     = blockIdx.z;
    const size_t headQK = (size_t)(b * H_HEADS + hh) * SEQ * DHD;
    const size_t headV  = (size_t)(b * H_HEADS + hh) * DHD * SEQ;
    const float  scale  = 0.125f;   // 1/sqrt(64)

    auto stageKV = [&](int buf, int kv0) {
#if HAVE_ASYNC_LDS
        #pragma unroll
        for (int i = 0; i < 2; ++i) {            // K: 256 x 16B chunks
            int idx = tid + i * 128;
            int row = idx >> 3, ch = (idx & 7) << 3;
            copy16_g2l(&sK[buf][row * LDKK + ch],
                       Kh + headQK + (size_t)(kv0 + row) * DHD + ch);
        }
        #pragma unroll
        for (int i = 0; i < 2; ++i) {            // V: 256 x 16B chunks
            int idx = tid + i * 128;
            int row = idx >> 2, ch = (idx & 3) << 3;
            copy16_g2l(&sV[buf][row * LDVK + ch],
                       Vt + headV + (size_t)row * SEQ + kv0 + ch);
        }
#else
        uint4 rk[2], rv[2];
        #pragma unroll
        for (int i = 0; i < 2; ++i) {
            int idx = tid + i * 128;
            int row = idx >> 3, ch = (idx & 7) << 3;
            rk[i] = *(const uint4*)(Kh + headQK + (size_t)(kv0 + row) * DHD + ch);
        }
        #pragma unroll
        for (int i = 0; i < 2; ++i) {
            int idx = tid + i * 128;
            int row = idx >> 2, ch = (idx & 3) << 3;
            rv[i] = *(const uint4*)(Vt + headV + (size_t)row * SEQ + kv0 + ch);
        }
        #pragma unroll
        for (int i = 0; i < 2; ++i) {
            int idx = tid + i * 128;
            int row = idx >> 3, ch = (idx & 7) << 3;
            *(uint4*)&sK[buf][row * LDKK + ch] = rk[i];
        }
        #pragma unroll
        for (int i = 0; i < 2; ++i) {
            int idx = tid + i * 128;
            int row = idx >> 2, ch = (idx & 3) << 3;
            *(uint4*)&sV[buf][row * LDVK + ch] = rv[i];
        }
#endif
    };

    // Q fragments (registers for the whole kv sweep)
    Frag16 aQ[2];
    #pragma unroll
    for (int c = 0; c < 2; ++c)
        #pragma unroll
        for (int v = 0; v < 8; ++v) {
            int k = c * 32 + ((v >> 2) << 4) + (lh << 3) + ((v & 3) << 1);
            aQ[c].u[v] = *(const unsigned*)(Qh + headQK + (size_t)(qbase + lm) * DHD + k);
        }

    float mrun[8], lrun[8];
    v8f   O[4];
    #pragma unroll
    for (int v = 0; v < 8; ++v) { mrun[v] = -1e30f; lrun[v] = 0.f; }
    #pragma unroll
    for (int c = 0; c < 4; ++c) O[c] = zero8();

    // prologue: stage first K/V tile
    stageKV(0, 0);
    wait_g2l();
    __syncthreads();

    int buf = 0;
    for (int kv0 = 0; kv0 < SEQ; kv0 += 32) {
        // stream next K/V tile while computing on this one
        if (kv0 + 32 < SEQ) stageKV(buf ^ 1, kv0 + 32);

        // ---- scores: two 16x16 kv tiles from sK[buf] ----
        v8f sc[2];
        #pragma unroll
        for (int t = 0; t < 2; ++t) {
            Frag16 bK[2];
            #pragma unroll
            for (int c = 0; c < 2; ++c)
                #pragma unroll
                for (int v = 0; v < 8; ++v) {
                    int k = c * 32 + ((v >> 2) << 4) + (lh << 3) + ((v & 3) << 1);
                    bK[c].u[v] = *(const unsigned*)&sK[buf][(t * 16 + lm) * LDKK + k];
                }
            v8f z = zero8();
            z = __builtin_amdgcn_wmma_f32_16x16x32_f16(false, aQ[0].v, false, bK[0].v,
                                                       (short)0, z, false, false);
            z = __builtin_amdgcn_wmma_f32_16x16x32_f16(false, aQ[1].v, false, bK[1].v,
                                                       (short)0, z, false, false);
            sc[t] = z;
        }

        // ---- online softmax update (rows r = v + 8*lh) ----
        #pragma unroll
        for (int v = 0; v < 8; ++v) {
            float s0 = sc[0][v] * scale;
            float s1 = sc[1][v] * scale;
            float x = fmaxf(s0, s1);
            x = fmaxf(x, __shfl_xor(x, 1));
            x = fmaxf(x, __shfl_xor(x, 2));
            x = fmaxf(x, __shfl_xor(x, 4));
            x = fmaxf(x, __shfl_xor(x, 8));
            float mn = fmaxf(mrun[v], x);
            float al = __expf(mrun[v] - mn);
            float p0 = __expf(s0 - mn);
            float p1 = __expf(s1 - mn);
            float rs = p0 + p1;
            rs += __shfl_xor(rs, 1);
            rs += __shfl_xor(rs, 2);
            rs += __shfl_xor(rs, 4);
            rs += __shfl_xor(rs, 8);
            lrun[v] = lrun[v] * al + rs;
            mrun[v] = mn;
            sc[0][v] = p0;
            sc[1][v] = p1;
            #pragma unroll
            for (int c = 0; c < 4; ++c) O[c][v] *= al;
        }

        // ---- transpose P (C-layout) -> A-layout via per-wave LDS slab ----
        #pragma unroll
        for (int t = 0; t < 2; ++t)
            #pragma unroll
            for (int v = 0; v < 8; ++v) {
                H1 cv; cv.h = (_Float16)sc[t][v];
                sP[wave][(v + (lh << 3)) * LPP + t * 16 + lm] = cv.s;
            }
        asm volatile("s_wait_dscnt 0x0" ::: "memory");   // cross-lane LDS hazard

        Frag16 aP;
        #pragma unroll
        for (int v = 0; v < 8; ++v) {
            int k = ((v >> 2) << 4) + (lh << 3) + ((v & 3) << 1);
            aP.u[v] = *(const unsigned*)&sP[wave][lm * LPP + k];
        }

        // ---- O += P * V  (4 dh tiles of 16, V fragments from sV[buf]) ----
        #pragma unroll
        for (int c = 0; c < 4; ++c) {
            Frag16 bV;
            #pragma unroll
            for (int v = 0; v < 8; ++v) {
                int k = ((v >> 2) << 4) + (lh << 3) + ((v & 3) << 1);
                bV.u[v] = *(const unsigned*)&sV[buf][(c * 16 + lm) * LDVK + k];
            }
            O[c] = __builtin_amdgcn_wmma_f32_16x16x32_f16(false, aP.v, false, bV.v,
                                                          (short)0, O[c], false, false);
        }

        wait_g2l();          // next K/V tile landed
        __syncthreads();     // all waves done with sK/sV[buf] (and sP WAR)
        buf ^= 1;
    }

    // ---- epilogue: divide by denom, store f16 (feeds final GEMM) ----
    #pragma unroll
    for (int c = 0; c < 4; ++c) {
        int dh = c * 16 + lm;
        #pragma unroll
        for (int v = 0; v < 8; ++v) {
            int row = qbase + v + (lh << 3);
            Out[(size_t)(b * SEQ + row) * DMODEL + hh * DHD + dh] =
                (_Float16)(O[c][v] / lrun[v]);
        }
    }
}

// ---------------------------------------------------------------------------
// Launcher
// ---------------------------------------------------------------------------
extern "C" void kernel_launch(void* const* d_in, const int* in_sizes, int n_in,
                              void* d_out, int out_size, void* d_ws, size_t ws_size,
                              hipStream_t stream) {
    (void)in_sizes; (void)n_in; (void)out_size; (void)ws_size;
    const float* x    = (const float*)d_in[0];
    const float* pe   = (const float*)d_in[1];
    const float* Wq   = (const float*)d_in[2];
    const float* bq   = (const float*)d_in[3];
    const float* Wk   = (const float*)d_in[4];
    const float* bk   = (const float*)d_in[5];
    const float* Wv   = (const float*)d_in[6];
    const float* bv   = (const float*)d_in[7];
    const float* qn_w = (const float*)d_in[8];
    const float* kn_w = (const float*)d_in[9];
    const float* Wo   = (const float*)d_in[10];
    const float* bo   = (const float*)d_in[11];
    float* out = (float*)d_out;

    const size_t nTok  = (size_t)BB * SEQ;           // 4096
    const size_t nElem = nTok * DMODEL;              // 4M
    const size_t nW    = (size_t)DMODEL * DMODEL;    // 1M

    // workspace carve-up
    float*    qf  = (float*)d_ws;                    // 3 x f32 activations
    float*    kf  = qf + nElem;
    float*    vf  = kf + nElem;
    _Float16* xh  = (_Float16*)(vf + nElem);         // f16 tensors
    _Float16* qh  = xh  + nElem;
    _Float16* kh  = qh  + nElem;
    _Float16* vt  = kh  + nElem;
    _Float16* aoh = vt  + nElem;
    _Float16* wqh = aoh + nElem;                     // f16 weights
    _Float16* wkh = wqh + nW;
    _Float16* wvh = wkh + nW;
    _Float16* woh = wvh + nW;

    // one-time f32 -> f16 conversions
    cvt_f32_to_f16<<<dim3((unsigned)(nElem / 4 / 256)), dim3(256), 0, stream>>>(x,  xh,  (int)(nElem / 4));
    cvt_f32_to_f16<<<dim3((unsigned)(nW / 4 / 256)),    dim3(256), 0, stream>>>(Wq, wqh, (int)(nW / 4));
    cvt_f32_to_f16<<<dim3((unsigned)(nW / 4 / 256)),    dim3(256), 0, stream>>>(Wk, wkh, (int)(nW / 4));
    cvt_f32_to_f16<<<dim3((unsigned)(nW / 4 / 256)),    dim3(256), 0, stream>>>(Wv, wvh, (int)(nW / 4));
    cvt_f32_to_f16<<<dim3((unsigned)(nW / 4 / 256)),    dim3(256), 0, stream>>>(Wo, woh, (int)(nW / 4));

    dim3 gGemm(DMODEL / GBN, (unsigned)(nTok / GBM)); // (16, 32)
    dim3 bGemm(128);

    // QKV projections
    gemm_h_xwT_bias<<<gGemm, bGemm, 0, stream>>>(xh, wqh, bq, qf, (int)nTok, DMODEL, DMODEL);
    gemm_h_xwT_bias<<<gGemm, bGemm, 0, stream>>>(xh, wkh, bk, kf, (int)nTok, DMODEL, DMODEL);
    gemm_h_xwT_bias<<<gGemm, bGemm, 0, stream>>>(xh, wvh, bv, vf, (int)nTok, DMODEL, DMODEL);

    // RMSNorm + RoPE -> f16 per-head tensors
    rmsnorm_rope_pack<<<dim3((unsigned)nTok), dim3(256), 0, stream>>>(qf, qn_w, pe, qh);
    rmsnorm_rope_pack<<<dim3((unsigned)nTok), dim3(256), 0, stream>>>(kf, kn_w, pe, kh);

    // V -> f16 transposed per head
    int total = (int)nElem;
    pack_v_t<<<dim3((total + 255) / 256), dim3(256), 0, stream>>>(vf, vt, total);

    // Attention (writes f16 activations for the output projection)
    dim3 gAttn(SEQ / 64, H_HEADS, BB);               // (32, 16, 2)
    attn_fwd<<<gAttn, dim3(128), 0, stream>>>(qh, kh, vt, aoh);

    // Output projection (f32 result to d_out)
    gemm_h_xwT_bias<<<gGemm, bGemm, 0, stream>>>(aoh, woh, bo, out, (int)nTok, DMODEL, DMODEL);
}
